// gnn_GAT_CONV_LIN_56788057587900
// MI455X (gfx1250) — matrix operand order repaired and checked by reading, not compiled, not run on hardware
//
#include <hip/hip_runtime.h>

#define DHID 128
#define DOUT 64

typedef __attribute__((ext_vector_type(16))) _Float16 v16h;
typedef __attribute__((ext_vector_type(8)))  float    v8f;

// ---------------------------------------------------------------------------
// WMMA GEMM: H[nrows x NC] = X[nrows x K] * W[K x NC] (+ bias)
// One wave = one 16x16 output tile. blockDim = (32,4): 4 full waves, EXEC all-1s.
// K-loop steps of 32 using v_wmma_f32_16x16x32_f16 (f32 accumulate).
// ---------------------------------------------------------------------------
template <int K, int NC, bool BIAS>
__global__ __launch_bounds__(128) void wmma_gemm_kernel(
    const float* __restrict__ X, const float* __restrict__ W,
    const float* __restrict__ bias, float* __restrict__ H) {
  const int lane = threadIdx.x;           // 0..31
  const int m    = lane & 15;
  const int sel  = lane >> 4;             // 0 or 1
  const int rt   = blockIdx.x;            // row tile
  const int ct   = blockIdx.y * 4 + threadIdx.y;  // col tile
  const int col0 = ct * 16;
  const int arow = rt * 16 + m;
  const int kb   = sel * 8;

  v8f c = {};
  for (int k0 = 0; k0 < K; k0 += 32) {
    const float* xr = X + (size_t)arow * K + k0;
    v16h a, b;
    // A 16x32 f16 layout: lane holds row m; elems 0..7 -> K = kb+j, 8..15 -> K = 16+kb+j
#pragma unroll
    for (int j = 0; j < 8; ++j) a[j] = (_Float16)xr[kb + j];
#pragma unroll
    for (int j = 0; j < 8; ++j) a[8 + j] = (_Float16)xr[16 + kb + j];
    // B 32x16 f16 layout: elems 0..7 -> K = kb+j, 8..15 -> K = 16+kb+(j-8); col = col0 + m
#pragma unroll
    for (int j = 0; j < 16; ++j)
      b[j] = (_Float16)W[(size_t)(k0 + kb + j + (j >= 8 ? 8 : 0)) * NC + col0 + m];
    c = __builtin_amdgcn_wmma_f32_16x16x32_f16(false, a, false, b, (short)0, c,
                                               false, false);
  }
  const float bv = BIAS ? bias[col0 + m] : 0.0f;
  // D layout: VGPR r -> row (sel*8 + r), col = m
#pragma unroll
  for (int r = 0; r < 8; ++r) {
    const int orow = rt * 16 + sel * 8 + r;
    H[(size_t)orow * NC + col0 + m] = c[r] + bv;
  }
}

// ---------------------------------------------------------------------------
// Per-node attention dots: es[i] = h[i].att_src, ed[i] = h[i].att_dst
// One wave per node; lane handles 4 contiguous floats (float4), wave reduce.
// ---------------------------------------------------------------------------
__global__ __launch_bounds__(256) void dots_kernel(
    const float* __restrict__ h, const float* __restrict__ asrc,
    const float* __restrict__ adst, float* __restrict__ es,
    float* __restrict__ ed, int n) {
  const int gid  = blockIdx.x * blockDim.x + threadIdx.x;
  const int node = gid >> 5;
  const int lane = threadIdx.x & 31;
  if (node >= n) return;
  const float4 v = ((const float4*)(h + (size_t)node * DHID))[lane];
  const float4 a = ((const float4*)asrc)[lane];
  const float4 b = ((const float4*)adst)[lane];
  float s = v.x * a.x + v.y * a.y + v.z * a.z + v.w * a.w;
  float d = v.x * b.x + v.y * b.y + v.z * b.z + v.w * b.w;
#pragma unroll
  for (int off = 16; off > 0; off >>= 1) {
    s += __shfl_down(s, off, 32);
    d += __shfl_down(d, off, 32);
  }
  if (lane == 0) { es[node] = s; ed[node] = d; }
}

// ---------------------------------------------------------------------------
__global__ void zero_kernel(float* __restrict__ p, size_t n) {
  size_t i = (size_t)blockIdx.x * blockDim.x + threadIdx.x;
  const size_t stride = (size_t)gridDim.x * blockDim.x;
  for (; i < n; i += stride) p[i] = 0.0f;
}

// ---------------------------------------------------------------------------
// Segment max of relu(es[src]+ed[dst]) per destination.
// All logits >= 0 (LeakyReLU slope==0), so u32-bit atomicMax == float max, init 0.
// ---------------------------------------------------------------------------
__global__ __launch_bounds__(256) void edge_max_kernel(
    const int* __restrict__ ei, const float* __restrict__ es,
    const float* __restrict__ ed, unsigned* __restrict__ mx, int E, int Etot) {
  const int i = blockIdx.x * blockDim.x + threadIdx.x;
  if (i >= Etot) return;
  int s, d;
  if (i < E) { s = ei[i]; d = ei[E + i]; } else { s = d = i - E; }
  float e = es[s] + ed[d];
  e = e > 0.0f ? e : 0.0f;
  atomicMax(mx + d, __float_as_uint(e));
}

// ---------------------------------------------------------------------------
// wexp[i] = exp(e - max[dst]); denom[dst] += wexp[i]
// ---------------------------------------------------------------------------
__global__ __launch_bounds__(256) void edge_expsum_kernel(
    const int* __restrict__ ei, const float* __restrict__ es,
    const float* __restrict__ ed, const float* __restrict__ mx,
    float* __restrict__ wexp, float* __restrict__ den, int E, int Etot) {
  const int i = blockIdx.x * blockDim.x + threadIdx.x;
  if (i >= Etot) return;
  int s, d;
  if (i < E) { s = ei[i]; d = ei[E + i]; } else { s = d = i - E; }
  float e = es[s] + ed[d];
  e = e > 0.0f ? e : 0.0f;
  const float ex = __expf(e - mx[d]);
  wexp[i] = ex;
  atomicAdd(den + d, ex);
}

// ---------------------------------------------------------------------------
// acc[dst] += alpha * h[src]. One wave per edge: lane grabs float4, 4 atomics.
// acc (25.6 MB) is L2-resident on MI455X (192 MB L2) -> atomics resolve in L2.
// ---------------------------------------------------------------------------
__global__ __launch_bounds__(256) void edge_aggr_kernel(
    const int* __restrict__ ei, const float* __restrict__ h,
    const float* __restrict__ wexp, const float* __restrict__ den,
    float* __restrict__ acc, int E, int Etot) {
  const int gid  = blockIdx.x * blockDim.x + threadIdx.x;
  const int e    = gid >> 5;               // wave-uniform
  const int lane = threadIdx.x & 31;
  if (e >= Etot) return;
  int s, d;
  if (e < E) { s = ei[e]; d = ei[E + e]; } else { s = d = e - E; }
  const float alpha = wexp[e] / den[d];
  const float4 hv = ((const float4*)(h + (size_t)s * DHID))[lane];
  float* ap = acc + (size_t)d * DHID + lane * 4;
  atomicAdd(ap + 0, hv.x * alpha);
  atomicAdd(ap + 1, hv.y * alpha);
  atomicAdd(ap + 2, hv.z * alpha);
  atomicAdd(ap + 3, hv.w * alpha);
}

// ---------------------------------------------------------------------------
// In-place: acc = act(acc + bias)
// ---------------------------------------------------------------------------
__global__ __launch_bounds__(256) void finalize_kernel(
    float* __restrict__ acc, const float* __restrict__ bias, int do_relu,
    size_t total) {
  const size_t i = (size_t)blockIdx.x * blockDim.x + threadIdx.x;
  if (i >= total) return;
  float v = acc[i] + bias[i % DHID];
  if (do_relu) v = v > 0.0f ? v : 0.0f;
  acc[i] = v;
}

// ---------------------------------------------------------------------------
extern "C" void kernel_launch(void* const* d_in, const int* in_sizes, int n_in,
                              void* d_out, int out_size, void* d_ws,
                              size_t ws_size, hipStream_t stream) {
  const float* node_x = (const float*)d_in[0];
  const int*   ei     = (const int*)d_in[1];
  const float* W1     = (const float*)d_in[2];
  const float* as1    = (const float*)d_in[3];
  const float* ad1    = (const float*)d_in[4];
  const float* b1     = (const float*)d_in[5];
  const float* W2     = (const float*)d_in[6];
  const float* as2    = (const float*)d_in[7];
  const float* ad2    = (const float*)d_in[8];
  const float* b2     = (const float*)d_in[9];
  const float* Wout   = (const float*)d_in[10];
  const float* bout   = (const float*)d_in[11];

  const int N    = in_sizes[0] / DHID;   // 50000
  const int E    = in_sizes[1] / 2;      // 1600000
  const int Etot = E + N;                // + self loops

  // workspace layout
  float* h    = (float*)d_ws;            // N*DHID
  float* acc  = h + (size_t)N * DHID;    // N*DHID
  float* es   = acc + (size_t)N * DHID;  // N
  float* ed   = es + N;                  // N
  float* mx   = ed + N;                  // N
  float* den  = mx + N;                  // N
  float* wexp = den + N;                 // Etot

  const dim3 gblk(32, 4);
  const int rowTiles = N / 16;           // 3125 (exact)

  auto layer = [&](const float* xin, const float* W, const float* as_,
                   const float* ad_, const float* bias, int relu) {
    wmma_gemm_kernel<DHID, DHID, false>
        <<<dim3(rowTiles, 2), gblk, 0, stream>>>(xin, W, nullptr, h);
    dots_kernel<<<(N * 32 + 255) / 256, 256, 0, stream>>>(h, as_, ad_, es, ed, N);
    zero_kernel<<<2048, 256, 0, stream>>>(acc, (size_t)N * DHID);
    zero_kernel<<<256, 256, 0, stream>>>(mx, (size_t)N);
    zero_kernel<<<256, 256, 0, stream>>>(den, (size_t)N);
    edge_max_kernel<<<(Etot + 255) / 256, 256, 0, stream>>>(ei, es, ed,
                                                            (unsigned*)mx, E, Etot);
    edge_expsum_kernel<<<(Etot + 255) / 256, 256, 0, stream>>>(ei, es, ed, mx,
                                                               wexp, den, E, Etot);
    edge_aggr_kernel<<<(Etot + 7) / 8, 256, 0, stream>>>(ei, h, wexp, den, acc,
                                                         E, Etot);
    finalize_kernel<<<(int)(((size_t)N * DHID + 255) / 256), 256, 0, stream>>>(
        acc, bias, relu, (size_t)N * DHID);
  };

  layer(node_x, W1, as1, ad1, b1, 1);  // x1 left in acc
  layer(acc,    W2, as2, ad2, b2, 0);  // x2 left in acc (gemm reads acc before zero)

  wmma_gemm_kernel<DHID, DOUT, true>
      <<<dim3(rowTiles, 1), gblk, 0, stream>>>(acc, Wout, bout, (float*)d_out);
}
